// RandomProjectionQuantizer_44074954391688
// MI455X (gfx1250) — matrix operand order repaired and checked by reading, not compiled, not run on hardware
//
#include <hip/hip_runtime.h>
#include <hip/hip_bf16.h>

typedef __attribute__((ext_vector_type(16))) _Float16 v16h;
typedef __attribute__((ext_vector_type(4)))  _Float16 v4h;
typedef __attribute__((ext_vector_type(8)))  float    v8f;

// Problem constants (b=4, n=2048, d=1024, h=8, e=16, c=2048)
#define D_DIM   1024
#define H_DIM   8
#define E_DIM   16
#define C_DIM   2048
#define KB_CNT  (D_DIM / 32)        // 32 K-blocks of 32 for the projection GEMM
#define CT_CNT  (C_DIM / 16)        // 128 codebook column tiles of 16

// ---------------------------------------------------------------------------
// Pack rand_projs [h,d,e] f32 -> per-lane WMMA B-fragments (f16).
// B layout for V_WMMA_F32_16X16X32_F16 (32x16, 8 VGPRs):
//   lanes 0-15 : N = lane,    elements j=0..7 -> K = j,     j=8..15 -> K = 16 + (j-8)
//   lanes 16-31: N = lane-16, elements j=0..7 -> K = 8 + j, j=8..15 -> K = 24 + (j-8)
// Wpack[h][kb][lane][j], 8*32*32*16 halves = 256KB.
// ---------------------------------------------------------------------------
__global__ void rpq_packW(const float* __restrict__ W, _Float16* __restrict__ Wpack) {
    int t = blockIdx.x * 256 + threadIdx.x;          // 131072 total
    int j    = t & 15;
    int lane = (t >> 4) & 31;
    int kb   = (t >> 9) & 31;
    int h    = t >> 14;
    int K = (j & 7) + 8 * (lane >> 4) + 16 * (j >> 3);
    int N = lane & 15;                               // e index
    int didx = kb * 32 + K;
    Wpack[t] = (_Float16)W[((size_t)(h * D_DIM + didx)) * E_DIM + N];
}

// ---------------------------------------------------------------------------
// L2-normalize codebook rows and pack as FULL B-fragments with K = e.
// K=16..31 (j=8..15 of every lane) are stored as zeros so the hot loop loads
// unconditionally -- no EXEC masking around the WMMA.
// CBpack[h][ct][lane][j] = [8][128][32][16] halves = 1MB.
// ---------------------------------------------------------------------------
__global__ void rpq_packCB(const float* __restrict__ cb, _Float16* __restrict__ CBpack) {
    int t = blockIdx.x * 256 + threadIdx.x;          // 16384 = h*c rows
    int h    = t >> 11;
    int code = t & 2047;
    int ct   = code >> 4;
    int lane = code & 15;
    const float* row = cb + (size_t)t * E_DIM;
    float v[E_DIM];
    float ss = 0.f;
#pragma unroll
    for (int e = 0; e < E_DIM; ++e) { v[e] = row[e]; ss += v[e] * v[e]; }
    float inv = rsqrtf(ss + 1e-12f);                 // L2_EPS
    _Float16* dst  = CBpack + ((size_t)((h * CT_CNT + ct) * 32 + lane)) * 16;
    _Float16* dstz = dst + 16 * 16;                  // lane+16: e=8..15 in j=0..7, zeros in j=8..15
#pragma unroll
    for (int e = 0; e < E_DIM; ++e) {
        dst[e]  = (_Float16)(e < 8 ? v[e] * inv : 0.f);
        dstz[e] = (_Float16)(e < 8 ? v[8 + e] * inv : 0.f);
    }
}

// ---------------------------------------------------------------------------
// Main fused kernel: LayerNorm -> projection WMMA -> similarity WMMA -> argmax
// Block = 16 tokens, 8 waves; wave w owns head w. 512 blocks cover 8192 tokens.
// ---------------------------------------------------------------------------
__global__ void __launch_bounds__(256)
rpq_main(const float* __restrict__ x, const _Float16* __restrict__ Wpack,
         const _Float16* __restrict__ CBpack, float* __restrict__ out) {
    // A-fragment pack for xn (16 tokens x 1024), already in WMMA A layout:
    //   lane<16 : M=lane,    j=0..7 -> K=0..7,  j=8..15 -> K=16..23
    //   lane>=16: M=lane-16, j=0..7 -> K=8..15, j=8..15 -> K=24..31
    __shared__ _Float16 apack[KB_CNT * 32 * 16];     // 32KB
    __shared__ float    projlds[H_DIM * 16 * 16];    // 8KB transpose bounce

    const int lane = threadIdx.x & 31;
    const int wave = threadIdx.x >> 5;               // == head id
    const int base = blockIdx.x * 16;                // token base

    // ---- Stage 1: LayerNorm 2 tokens per wave, write f16 A-fragments ----
#pragma unroll
    for (int tt = 0; tt < 2; ++tt) {
        const int m = wave * 2 + tt;                 // token row within tile
        const float4* row = (const float4*)(x + (size_t)(base + m) * D_DIM);
        float4 v[8];
        float s = 0.f, ss = 0.f;
#pragma unroll
        for (int i = 0; i < 8; ++i) {
            v[i] = row[lane + 32 * i];
            s  += v[i].x + v[i].y + v[i].z + v[i].w;
            ss += v[i].x * v[i].x + v[i].y * v[i].y + v[i].z * v[i].z + v[i].w * v[i].w;
        }
#pragma unroll
        for (int mask = 16; mask >= 1; mask >>= 1) {
            s  += __shfl_xor(s,  mask, 32);
            ss += __shfl_xor(ss, mask, 32);
        }
        const float mean = s * (1.f / D_DIM);
        const float var  = ss * (1.f / D_DIM) - mean * mean;
        const float inv  = rsqrtf(var + 1e-5f);      // LayerNorm EPS
        // The 4 elements of each float4 map to the same dest lane-row with 4
        // consecutive j (k0 is 4-aligned), so pack them into one b64 DS store.
#pragma unroll
        for (int i = 0; i < 8; ++i) {
            const int k0 = (lane + 32 * i) * 4;
            const int kb = k0 >> 5, ko = k0 & 31;
            const int hl = (ko >> 3) & 1;                     // lane half
            const int dl = m + 16 * hl;
            const int j  = (ko & 7) + 8 * ((ko >> 4) & 1);    // 4-aligned
            v4h q;
            q[0] = (_Float16)((v[i].x - mean) * inv);
            q[1] = (_Float16)((v[i].y - mean) * inv);
            q[2] = (_Float16)((v[i].z - mean) * inv);
            q[3] = (_Float16)((v[i].w - mean) * inv);
            *(v4h*)&apack[((kb * 32 + dl) << 4) + j] = q;
        }
    }
    __syncthreads();

    // ---- Stage 2: projection GEMM, head = wave: proj[16x16] += A(16x32)B(32x16)
    v8f acc = {};
    const _Float16* wp = Wpack + (size_t)wave * KB_CNT * 32 * 16;
#pragma unroll 4
    for (int kb = 0; kb < KB_CNT; ++kb) {
        v16h a = *(const v16h*)&apack[(kb * 32 + lane) << 4];
        v16h b = *(const v16h*)&wp[(kb * 32 + lane) << 4];
        acc = __builtin_amdgcn_wmma_f32_16x16x32_f16(
            false, a, false, b, (short)0, acc, false, false);
    }

    // ---- Stage 3: transpose accumulator into next A-fragment via LDS ----
    // C/D layout: lane l, vgpr r -> (M = r + 8*(l>>4), N = l&15)
    {
        const int N   = lane & 15;
        const int mhi = (lane >> 4) * 8;
#pragma unroll
        for (int r = 0; r < 8; ++r)
            projlds[(wave * 16 + (r + mhi)) * 16 + N] = acc[r];
    }
    __syncthreads();

    v16h a2;
    {
        const int mrow  = lane & 15;                  // M
        const int ebase = (lane >> 4) * 8;            // K base (e)
#pragma unroll
        for (int j = 0; j < 8; ++j)
            a2[j] = (_Float16)projlds[(wave * 16 + mrow) * 16 + ebase + j];
#pragma unroll
        for (int j = 8; j < 16; ++j) a2[j] = (_Float16)0.f;   // K=16..31 pad
    }

    // ---- Stage 4: similarity GEMM vs codebook + running argmax ----
    float rmax[8];
    int   ridx[8];
#pragma unroll
    for (int r = 0; r < 8; ++r) { rmax[r] = -__builtin_inff(); ridx[r] = 0; }

    const v8f zero_c = {};
    const _Float16* cbp = CBpack + (size_t)wave * CT_CNT * 32 * 16;
#pragma unroll 4
    for (int ct = 0; ct < CT_CNT; ++ct) {
        v16h b2 = *(const v16h*)&cbp[(ct * 32 + lane) << 4];  // K-pad pre-stored
        v8f d = __builtin_amdgcn_wmma_f32_16x16x32_f16(
            false, a2, false, b2, (short)0, zero_c, false, false);
        const int code = ct * 16 + (lane & 15);       // this lane's column N
        // Branchless: v_max_f32 + v_cndmask; strict > keeps earliest ct on ties.
#pragma unroll
        for (int r = 0; r < 8; ++r) {
            const bool gt = d[r] > rmax[r];
            rmax[r] = fmaxf(rmax[r], d[r]);
            ridx[r] = gt ? code : ridx[r];
        }
    }

    // ---- Stage 5: reduce (max, argmax) across the 16 columns, write out ----
#pragma unroll
    for (int r = 0; r < 8; ++r) {
        float v = rmax[r];
        int   i = ridx[r];
#pragma unroll
        for (int mask = 1; mask < 16; mask <<= 1) {   // stays within 16-lane half
            float v2 = __shfl_xor(v, mask, 32);
            int   i2 = __shfl_xor(i, mask, 32);
            const bool take = (v2 > v) || (v2 == v && i2 < i);  // first-max
            v = take ? v2 : v;
            i = take ? i2 : i;
        }
        if ((lane & 15) == 0) {
            const int M = r + 8 * (lane >> 4);
            out[(size_t)(base + M) * H_DIM + wave] = (float)i;
        }
    }
}

extern "C" void kernel_launch(void* const* d_in, const int* in_sizes, int n_in,
                              void* d_out, int out_size, void* d_ws, size_t ws_size,
                              hipStream_t stream) {
    const float* x  = (const float*)d_in[0];   // [4,2048,1024]
    const float* W  = (const float*)d_in[1];   // [8,1024,16]
    const float* cb = (const float*)d_in[2];   // [8,2048,16]
    float* out = (float*)d_out;                // [4,2048,8] indices as float

    _Float16* Wpack  = (_Float16*)d_ws;                              // 256KB
    _Float16* CBpack = (_Float16*)((char*)d_ws + 256 * 1024);        // 1MB

    rpq_packW <<<512, 256, 0, stream>>>(W,  Wpack);
    rpq_packCB<<< 64, 256, 0, stream>>>(cb, CBpack);
    rpq_main  <<<512, 256, 0, stream>>>(x, Wpack, CBpack, out);
}
